// GAT_47476568490430
// MI455X (gfx1250) — matrix operand order repaired and checked by reading, not compiled, not run on hardware
//
#include <hip/hip_runtime.h>
#include <hip/hip_bf16.h>
#include <math.h>

// ---------------------------------------------------------------------------
// Types for CDNA5 (gfx1250) WMMA
// ---------------------------------------------------------------------------
typedef __bf16 bf16_t;
typedef bf16_t v16bf __attribute__((ext_vector_type(16)));
typedef float  v8f   __attribute__((ext_vector_type(8)));
typedef unsigned int u32x4 __attribute__((ext_vector_type(4)));

#define GAT_N_NODES 50000
#define GAT_N_EDGES 800000
#define GAT_IN_CH   512
#define GAT_HID     256
#define GAT_HEADS   8
#define GAT_OUT_CH  64

// ---------------------------------------------------------------------------
// Elementwise helpers
// ---------------------------------------------------------------------------
__global__ void fill_f32_kernel(float* __restrict__ p, float v, int n) {
    int i = blockIdx.x * blockDim.x + threadIdx.x;
    if (i < n) p[i] = v;
}

__global__ void cvt_f32_to_bf16_kernel(const float* __restrict__ in,
                                       bf16_t* __restrict__ out, int n) {
    int i = blockIdx.x * blockDim.x + threadIdx.x;
    if (i < n) out[i] = (bf16_t)in[i];
}

// W fp32 [K][N] -> Wt bf16 [N][K]  (so GEMM B-operand is K-contiguous)
__global__ void cvt_transpose_w_kernel(const float* __restrict__ W,
                                       bf16_t* __restrict__ Wt, int K, int N) {
    int i = blockIdx.x * blockDim.x + threadIdx.x;
    if (i >= K * N) return;
    int k = i / N;
    int n = i - k * N;
    Wt[(size_t)n * K + k] = (bf16_t)W[i];
}

// ---------------------------------------------------------------------------
// gfx1250 async global->LDS DMA (ASYNCcnt-tracked), per CDNA5 ISA.
// LDS destination address = per-lane VGPR (low 32 bits of flat shared addr).
// ---------------------------------------------------------------------------
__device__ __forceinline__ void async_load_lds_b128(unsigned lds_off,
                                                    const void* gptr) {
    unsigned long long ga = (unsigned long long)(uintptr_t)gptr;
    asm volatile("global_load_async_to_lds_b128 %0, %1, off"
                 :: "v"(lds_off), "v"(ga)
                 : "memory");
}

__device__ __forceinline__ void wait_asynccnt0() {
    asm volatile("s_wait_asynccnt 0x0" ::: "memory");
}

__device__ __forceinline__ unsigned lds_offset_u32(const void* shared_ptr) {
    // LDS offset is the low 32 bits of the flat shared address
    return (unsigned)(uintptr_t)shared_ptr;
}

// ---------------------------------------------------------------------------
// WMMA bf16 GEMM: C[M][N] = A[M][K] * Bt[N][K]^T   (fp32 accumulate)
// Block tile 128x64, BK=32, 256 threads = 8 wave32; wave owns 32x32 quadrant.
// Double-buffered LDS filled by async DMA; one barrier per K-step.
// ---------------------------------------------------------------------------
#define BM 128
#define BN 64
#define BK 32
#define LDS_STRIDE 40   // bf16 units: 80B row stride -> conflict-free b128 reads

union AFragU { u32x4 u[2]; v16bf v; };

__global__ __launch_bounds__(256)
void gemm_bf16_wmma_kernel(const bf16_t* __restrict__ A,
                           const bf16_t* __restrict__ Bt,
                           float* __restrict__ C,
                           int M, int N, int K) {
    __shared__ __align__(16) bf16_t As[2][BM * LDS_STRIDE];
    __shared__ __align__(16) bf16_t Bs[2][BN * LDS_STRIDE];

    const int tid   = threadIdx.x;
    const int lane  = tid & 31;
    const int wave  = tid >> 5;     // 0..7
    const int wm    = wave & 3;     // wave-row (4)
    const int wn    = wave >> 2;    // wave-col (2)
    const int m0    = blockIdx.x * BM;
    const int n0    = blockIdx.y * BN;
    const int lhalf = lane >> 4;    // 0 or 1
    const int l15   = lane & 15;

    v8f acc[2][2] = {};

    // ---- staging assignment: thread -> (row, 16B chunk) ----
    const int part = tid & 3;       // 8 bf16 (16B) chunk within a 32-elem row
    const int lrow = tid >> 2;      // 0..63

    // Clamp out-of-range rows to M-1: loads stay unconditional (no exec
    // diamonds); garbage rows only feed C rows that are never stored.
    int arow0 = m0 + lrow;        if (arow0 >= M) arow0 = M - 1;
    int arow1 = m0 + lrow + 64;   if (arow1 >= M) arow1 = M - 1;

    const bf16_t* gA0 = A  + (size_t)arow0 * K + part * 8;
    const bf16_t* gA1 = A  + (size_t)arow1 * K + part * 8;
    const bf16_t* gB  = Bt + (size_t)(n0 + lrow) * K + part * 8;

    unsigned ldsA0[2], ldsA1[2], ldsB[2];
    #pragma unroll
    for (int b = 0; b < 2; ++b) {
        ldsA0[b] = lds_offset_u32(&As[b][lrow * LDS_STRIDE + part * 8]);
        ldsA1[b] = lds_offset_u32(&As[b][(lrow + 64) * LDS_STRIDE + part * 8]);
        ldsB[b]  = lds_offset_u32(&Bs[b][lrow * LDS_STRIDE + part * 8]);
    }

    // ---- prologue: async-fill buffer 0 with K-tile 0 ----
    async_load_lds_b128(ldsA0[0], gA0);
    async_load_lds_b128(ldsA1[0], gA1);
    async_load_lds_b128(ldsB[0],  gB);

    const int nT = K / BK;
    for (int t = 0; t < nT; ++t) {
        const int cur = t & 1;
        wait_asynccnt0();       // my DMA fills of buf[cur] have landed
        __syncthreads();        // everyone's fills landed; buf[cur^1] free
        if (t + 1 < nT) {
            const int nxt = cur ^ 1;
            const size_t ko = (size_t)(t + 1) * BK;
            async_load_lds_b128(ldsA0[nxt], gA0 + ko);
            async_load_lds_b128(ldsA1[nxt], gA1 + ko);
            async_load_lds_b128(ldsB[nxt],  gB + ko);
        }

        // ---- build fragments (layout per CDNA5 ISA 16-bit A/B tables) ----
        AFragU afr[2], bfr[2];
        #pragma unroll
        for (int i = 0; i < 2; ++i) {
            // A: lane<16 -> row M=l15, K 0-7 (u[0]) & 16-23 (u[1]); lane>=16 -> K 8-15 & 24-31
            const bf16_t* pa = &As[cur][(wm * 32 + i * 16 + l15) * LDS_STRIDE + lhalf * 8];
            afr[i].u[0] = *(const u32x4*)(pa);
            afr[i].u[1] = *(const u32x4*)(pa + 16);
            // B: lane<16 -> col N=l15, K 0-7 (u[0]) & 16-23 (u[1]); lane>=16 -> K 8-15 & 24-31
            const bf16_t* pb = &Bs[cur][(wn * 32 + i * 16 + l15) * LDS_STRIDE + lhalf * 8];
            bfr[i].u[0] = *(const u32x4*)(pb);
            bfr[i].u[1] = *(const u32x4*)(pb + 16);
        }

        // ---- four 16x16x32 WMMAs per wave per K step ----
        #pragma unroll
        for (int i = 0; i < 2; ++i)
            #pragma unroll
            for (int j = 0; j < 2; ++j)
                acc[i][j] = __builtin_amdgcn_wmma_f32_16x16x32_bf16(
                    /*neg_a=*/false, afr[i].v, /*neg_b=*/false, bfr[j].v,
                    /*c_mod=*/(short)0, acc[i][j],
                    /*reuse_a=*/false, /*reuse_b=*/false);
    }

    // ---- store C (16x16 f32 D layout: vgpr r -> M=r or r+8; lane -> N) ----
    #pragma unroll
    for (int i = 0; i < 2; ++i)
        #pragma unroll
        for (int j = 0; j < 2; ++j) {
            int col = n0 + wn * 32 + j * 16 + l15;
            #pragma unroll
            for (int r = 0; r < 8; ++r) {
                int row = m0 + wm * 32 + i * 16 + r + 8 * lhalf;
                if (row < M) C[(size_t)row * N + col] = acc[i][j][r];
            }
        }
}

// ---------------------------------------------------------------------------
// Attention-score node reduction: alpha_src/dst[n,h] = sum_c h[n,h,c]*a[h,c]
// ---------------------------------------------------------------------------
__global__ void alpha_kernel(const float* __restrict__ h,
                             const float* __restrict__ a_src,
                             const float* __restrict__ a_dst,
                             int N, int H, int C,
                             float* __restrict__ asrc, float* __restrict__ adst) {
    int idx = blockIdx.x * blockDim.x + threadIdx.x;
    if (idx >= N * H) return;
    int n  = idx / H;
    int hh = idx - n * H;
    const float* hp = h + (size_t)n * H * C + hh * C;
    float ss = 0.f, sd = 0.f;
    for (int c = 0; c < C; ++c) {
        float v = hp[c];
        ss += v * a_src[hh * C + c];
        sd += v * a_dst[hh * C + c];
    }
    asrc[idx] = ss;
    adst[idx] = sd;
}

// ---------------------------------------------------------------------------
// Edge passes (self-loops appended virtually: edge e >= E -> (e-E, e-E))
// ---------------------------------------------------------------------------
__device__ inline void edge_sd(const int* __restrict__ ei, int E, int e, int& s, int& d) {
    if (e < E) { s = ei[e]; d = ei[E + e]; }
    else       { s = e - E; d = s; }
}

__device__ inline float edge_e(const float* __restrict__ asrc,
                               const float* __restrict__ adst,
                               int s, int d, int H, int h) {
    float v = asrc[s * H + h] + adst[d * H + h];
    return v > 0.f ? v : 0.2f * v;   // leaky_relu(0.2)
}

__device__ inline void atomicMaxF32(float* addr, float val) {
    int* ia = (int*)addr;
    int old = *ia;
    while (__int_as_float(old) < val) {
        int assumed = old;
        old = atomicCAS(ia, assumed, __float_as_int(val));
        if (old == assumed) break;
    }
}

__global__ void edge_max_kernel(const float* __restrict__ asrc,
                                const float* __restrict__ adst,
                                const int* __restrict__ ei,
                                int E, int Etot, int H,
                                float* __restrict__ mbuf) {
    int idx = blockIdx.x * blockDim.x + threadIdx.x;
    if (idx >= Etot * H) return;
    int e = idx / H, h = idx - e * H;
    int s, d; edge_sd(ei, E, e, s, d);
    atomicMaxF32(&mbuf[d * H + h], edge_e(asrc, adst, s, d, H, h));
}

__global__ void edge_sum_kernel(const float* __restrict__ asrc,
                                const float* __restrict__ adst,
                                const int* __restrict__ ei,
                                int E, int Etot, int H,
                                const float* __restrict__ mbuf,
                                float* __restrict__ ssum) {
    int idx = blockIdx.x * blockDim.x + threadIdx.x;
    if (idx >= Etot * H) return;
    int e = idx / H, h = idx - e * H;
    int s, d; edge_sd(ei, E, e, s, d);
    float w = __expf(edge_e(asrc, adst, s, d, H, h) - mbuf[d * H + h]);
    atomicAdd(&ssum[d * H + h], w);
}

__global__ void edge_scatter_kernel(const float* __restrict__ asrc,
                                    const float* __restrict__ adst,
                                    const int* __restrict__ ei,
                                    int E, int Etot, int H, int C,
                                    const float* __restrict__ mbuf,
                                    const float* __restrict__ ssum,
                                    const float* __restrict__ h,
                                    float* __restrict__ agg) {
    int idx = blockIdx.x * blockDim.x + threadIdx.x;
    if (idx >= Etot * H) return;
    int e = idx / H, hh = idx - e * H;
    int s, d; edge_sd(ei, E, e, s, d);
    float w = __expf(edge_e(asrc, adst, s, d, H, hh) - mbuf[d * H + hh]);
    float alpha = w / (ssum[d * H + hh] + 1e-16f);
    const float* hp = h   + (size_t)s * H * C + hh * C;
    float*       op = agg + (size_t)d * H * C + hh * C;
    for (int c = 0; c < C; ++c)
        atomicAdd(&op[c], alpha * hp[c]);
}

// bias + ELU + bf16 cast (feed into next layer's GEMM)
__global__ void post_elu_bf16_kernel(const float* __restrict__ agg,
                                     const float* __restrict__ bias,
                                     int N, int F, bf16_t* __restrict__ out) {
    int i = blockIdx.x * blockDim.x + threadIdx.x;
    if (i >= N * F) return;
    float v = agg[i] + bias[i % F];
    v = v > 0.f ? v : (__expf(v) - 1.f);
    out[i] = (bf16_t)v;
}

// final: out = log_softmax(agg + b2) over F=64 channels
__global__ void final_logsoftmax_kernel(const float* __restrict__ agg,
                                        const float* __restrict__ b2,
                                        int N, int F, float* __restrict__ out) {
    int n = blockIdx.x * blockDim.x + threadIdx.x;
    if (n >= N) return;
    const float* ap = agg + (size_t)n * F;
    float mx = -1e30f;
    for (int c = 0; c < F; ++c) {
        float x = ap[c] + b2[c];
        mx = x > mx ? x : mx;
    }
    float se = 0.f;
    for (int c = 0; c < F; ++c) se += __expf(ap[c] + b2[c] - mx);
    float lse = logf(se);
    for (int c = 0; c < F; ++c) out[(size_t)n * F + c] = ap[c] + b2[c] - mx - lse;
}

// ---------------------------------------------------------------------------
// Launch
// ---------------------------------------------------------------------------
static inline int ceil_div(int a, int b) { return (a + b - 1) / b; }

extern "C" void kernel_launch(void* const* d_in, const int* in_sizes, int n_in,
                              void* d_out, int out_size, void* d_ws, size_t ws_size,
                              hipStream_t stream) {
    const float* x      = (const float*)d_in[0];
    const int*   ei     = (const int*)  d_in[1];
    const float* W0     = (const float*)d_in[2];
    const float* a_src0 = (const float*)d_in[3];
    const float* a_dst0 = (const float*)d_in[4];
    const float* b0     = (const float*)d_in[5];
    const float* W1     = (const float*)d_in[6];
    const float* a_src1 = (const float*)d_in[7];
    const float* a_dst1 = (const float*)d_in[8];
    const float* b1     = (const float*)d_in[9];
    const float* W2     = (const float*)d_in[10];
    const float* a_src2 = (const float*)d_in[11];
    const float* a_dst2 = (const float*)d_in[12];
    const float* b2     = (const float*)d_in[13];

    const int N    = in_sizes[0] / GAT_IN_CH;   // 50000
    const int E    = in_sizes[1] / 2;           // 800000
    const int Etot = E + N;                     // + self loops

    // ---- carve workspace ----
    char* p = (char*)d_ws;
    auto alloc = [&](size_t bytes) -> void* {
        void* r = (void*)p;
        p += (bytes + 255) & ~(size_t)255;
        return r;
    };
    bf16_t* xb    = (bf16_t*)alloc((size_t)N * GAT_IN_CH * sizeof(bf16_t));
    bf16_t* W0t   = (bf16_t*)alloc((size_t)GAT_IN_CH * GAT_HID * sizeof(bf16_t));
    bf16_t* W1t   = (bf16_t*)alloc((size_t)GAT_HID * GAT_HID * sizeof(bf16_t));
    bf16_t* W2t   = (bf16_t*)alloc((size_t)GAT_HID * GAT_OUT_CH * sizeof(bf16_t));
    float*  h     = (float*) alloc((size_t)N * GAT_HID * sizeof(float));
    float*  agg   = (float*) alloc((size_t)N * GAT_HID * sizeof(float));
    bf16_t* featb = (bf16_t*)alloc((size_t)N * GAT_HID * sizeof(bf16_t));
    float*  asrc  = (float*) alloc((size_t)N * GAT_HEADS * sizeof(float));
    float*  adst  = (float*) alloc((size_t)N * GAT_HEADS * sizeof(float));
    float*  mbuf  = (float*) alloc((size_t)N * GAT_HEADS * sizeof(float));
    float*  ssum  = (float*) alloc((size_t)N * GAT_HEADS * sizeof(float));

    const int TB = 256;

    // ---- precision conversion ----
    cvt_f32_to_bf16_kernel<<<ceil_div(N * GAT_IN_CH, TB), TB, 0, stream>>>(x, xb, N * GAT_IN_CH);
    cvt_transpose_w_kernel<<<ceil_div(GAT_IN_CH * GAT_HID, TB), TB, 0, stream>>>(W0, W0t, GAT_IN_CH, GAT_HID);
    cvt_transpose_w_kernel<<<ceil_div(GAT_HID * GAT_HID, TB), TB, 0, stream>>>(W1, W1t, GAT_HID, GAT_HID);
    cvt_transpose_w_kernel<<<ceil_div(GAT_HID * GAT_OUT_CH, TB), TB, 0, stream>>>(W2, W2t, GAT_HID, GAT_OUT_CH);

    auto run_gemm = [&](const bf16_t* A, const bf16_t* Bt, float* C, int M, int Nn, int K) {
        dim3 grid(ceil_div(M, BM), Nn / BN);
        gemm_bf16_wmma_kernel<<<grid, 256, 0, stream>>>(A, Bt, C, M, Nn, K);
    };

    auto run_attention = [&](const float* hmat, const float* avs, const float* avd, int H, int C) {
        int NH = N * H;
        fill_f32_kernel<<<ceil_div(NH, TB), TB, 0, stream>>>(mbuf, -1e30f, NH);
        fill_f32_kernel<<<ceil_div(NH, TB), TB, 0, stream>>>(ssum, 0.f, NH);
        fill_f32_kernel<<<ceil_div(N * H * C, TB), TB, 0, stream>>>(agg, 0.f, N * H * C);
        alpha_kernel<<<ceil_div(NH, TB), TB, 0, stream>>>(hmat, avs, avd, N, H, C, asrc, adst);
        int EH = Etot * H;
        edge_max_kernel    <<<ceil_div(EH, TB), TB, 0, stream>>>(asrc, adst, ei, E, Etot, H, mbuf);
        edge_sum_kernel    <<<ceil_div(EH, TB), TB, 0, stream>>>(asrc, adst, ei, E, Etot, H, mbuf, ssum);
        edge_scatter_kernel<<<ceil_div(EH, TB), TB, 0, stream>>>(asrc, adst, ei, E, Etot, H, C, mbuf, ssum, hmat, agg);
    };

    // ---- layer 0: in 512 -> 8x32 concat ----
    run_gemm(xb, W0t, h, N, GAT_HID, GAT_IN_CH);
    run_attention(h, a_src0, a_dst0, GAT_HEADS, GAT_HID / GAT_HEADS);
    post_elu_bf16_kernel<<<ceil_div(N * GAT_HID, TB), TB, 0, stream>>>(agg, b0, N, GAT_HID, featb);

    // ---- layer 1: 256 -> 8x32 concat ----
    run_gemm(featb, W1t, h, N, GAT_HID, GAT_HID);
    run_attention(h, a_src1, a_dst1, GAT_HEADS, GAT_HID / GAT_HEADS);
    post_elu_bf16_kernel<<<ceil_div(N * GAT_HID, TB), TB, 0, stream>>>(agg, b1, N, GAT_HID, featb);

    // ---- layer 2: 256 -> 1x64, mean over 1 head == identity ----
    run_gemm(featb, W2t, h, N, GAT_OUT_CH, GAT_HID);
    run_attention(h, a_src2, a_dst2, 1, GAT_OUT_CH);
    final_logsoftmax_kernel<<<ceil_div(N, TB), TB, 0, stream>>>(agg, b2, N, GAT_OUT_CH, (float*)d_out);
}
